// LiteMLA_29128468202044
// MI455X (gfx1250) — compile-verified, matrix operand order checked by reading, not run on hardware
//
#include <hip/hip_runtime.h>
#include <hip/hip_bf16.h>

// LiteMLA fused pipeline for MI455X (gfx1250), wave32 + WMMA bf16, bf16 intermediates.
//
// Workspace layout (bytes), total 407,388,160:
//   [0          , 100663296)  qkv_bf : (16,768,4096) bf16
//   [100663296  , 201326592)  ms2_bf : (16,768,4096) bf16
//   [201326592  , 205537280)  vk_bf  : (32,257,256)  bf16
//   [205537280  , 340279296)  raw    : (32,257,4096) f32  (pre-division, fp32 denominator)
//   [340279296  , 407388160)  attn_bf: (16,512,4096) bf16

typedef __attribute__((ext_vector_type(16))) __bf16   v16bf;
typedef __attribute__((ext_vector_type(8)))  __bf16   v8bf;
typedef __attribute__((ext_vector_type(8)))  float    v8f;
typedef __attribute__((ext_vector_type(4)))  float    v4f;
typedef __attribute__((ext_vector_type(4)))  unsigned v4u;

__device__ __forceinline__ unsigned short f2bf_bits(float f) {
  unsigned u = __builtin_bit_cast(unsigned, f);
  u += 0x7FFFu + ((u >> 16) & 1u);            // round-to-nearest-even
  return (unsigned short)(u >> 16);
}
__device__ __forceinline__ float bf2f(unsigned short s) {
  return __builtin_bit_cast(float, ((unsigned)s) << 16);
}

// relu on a pair of packed bf16: zero any half with the sign bit set
__device__ __forceinline__ unsigned relu_pk(unsigned u) {
  return u & ((((~u) & 0x80008000u) >> 15) * 0xFFFFu);
}

// load 8 consecutive elements as 4 packed bf16-pairs (vectorized 16B loads)
__device__ __forceinline__ void load8_bf(const float* p, unsigned* pk) {
  v4f lo = *(const v4f*)p;
  v4f hi = *(const v4f*)(p + 4);
  pk[0] = (unsigned)f2bf_bits(lo[0]) | ((unsigned)f2bf_bits(lo[1]) << 16);
  pk[1] = (unsigned)f2bf_bits(lo[2]) | ((unsigned)f2bf_bits(lo[3]) << 16);
  pk[2] = (unsigned)f2bf_bits(hi[0]) | ((unsigned)f2bf_bits(hi[1]) << 16);
  pk[3] = (unsigned)f2bf_bits(hi[2]) | ((unsigned)f2bf_bits(hi[3]) << 16);
}
__device__ __forceinline__ void load8_bf(const unsigned short* p, unsigned* pk) {
  v4u t = *(const v4u*)p;
  pk[0] = t[0]; pk[1] = t[1]; pk[2] = t[2]; pk[3] = t[3];
}
__device__ __forceinline__ unsigned short elem_bf(const float* p)          { return f2bf_bits(*p); }
__device__ __forceinline__ unsigned short elem_bf(const unsigned short* p) { return *p; }

__device__ __forceinline__ void store_c(float* p, float v)          { *p = v; }
__device__ __forceinline__ void store_c(unsigned short* p, float v) { *p = f2bf_bits(v); }

// Fragment load: with k-contiguous LDS rows, a lane's fragment = two contiguous
// 16B chunks at k = lh*8 and k = 16 + lh*8 (ISA 16-bit A/B VGPR layout:
// k(j) = ((j>>3)<<4) + lh*8 + (j&7)).  Lowers to 2x ds_load_b128.
struct BfPair { v8bf lo, hi; };
__device__ __forceinline__ v16bf load_frag(const unsigned short* rowbase, int lh) {
  BfPair p;
  p.lo = *(const v8bf*)(rowbase + lh * 8);
  p.hi = *(const v8bf*)(rowbase + 16 + lh * 8);
  return __builtin_bit_cast(v16bf, p);
}

// Generic 64x64-tile WMMA GEMM: C = A(MxK) * B(KxN), bf16 compute / f32 accum.
// All shapes/leading dims are compile-time: strided element offsets fold into
// global_load immediate offsets (no 64-bit address multiplies in the loop).
template<typename TA, typename TB, typename TC,
         int M, int N, int K, int LDA, int LDB, int LDC,
         bool A_PAIR, bool A_ONES, bool M_GUARD,
         bool B_PAIR, bool B_RELU, bool B_TRANS, bool DO_BN>
__global__ __launch_bounds__(256) void wmma_gemm(
    const TA* __restrict__ A0, const TA* __restrict__ A1, long aStride, long aOff,
    const TB* __restrict__ B0, const TB* __restrict__ B1, long bStride, long bOff,
    TC* __restrict__ C, long cStride,
    const float* __restrict__ bn_g, const float* __restrict__ bn_b,
    const float* __restrict__ bn_mn, const float* __restrict__ bn_vr)
{
  __shared__ unsigned short As[2][64][40];   // [buf][m][k], 16B-aligned chunks
  __shared__ unsigned short Bs[2][64][40];   // [buf][n][k]  (B stored transposed)

  const int z = blockIdx.z;
  const TA* Ap;
  const TB* Bp;
  if (A_PAIR) Ap = ((z & 1) ? A1 : A0) + (long)(z >> 1) * aStride + aOff;
  else        Ap = A0 + (long)z * aStride + aOff;
  if (B_PAIR) Bp = ((z & 1) ? B1 : B0) + (long)(z >> 1) * bStride + bOff;
  else        Bp = B0 + (long)z * bStride + bOff;
  TC* Cp = C + (long)z * cStride;

  const int t     = threadIdx.x;
  const int mBase = blockIdx.y * 64;
  const int nBase = blockIdx.x * 64;

  const int lane = t & 31;
  const int wave = t >> 5;
  const int wm = wave & 3;         // 4 row-slabs of 16
  const int wn = wave >> 2;        // 2 col-slabs of 32
  const int lm = lane & 15;
  const int lh = lane >> 4;

  // staging assignments (uniform per thread; guards are per-row, branch-free inner)
  const int aRow = t >> 2;         // 0..63
  const int aKc  = (t & 3) << 3;   // 0,8,16,24
  const int bRow = t & 63;         // n within tile (coalesced across threads)
  const int bKc  = (t >> 6) << 3;  // 0,8,16,24

  // hoisted base pointers, advanced by constant strides per stage
  const int  mg   = mBase + aRow;
  const bool aInB = !M_GUARD || (mg < (A_ONES ? M - 1 : M));
  const TA* aPtr = Ap + (long)mg * LDA + aKc;
  const TB* bPtr = B_TRANS ? (Bp + (long)(nBase + bRow) * LDB + bKc)
                           : (Bp + (long)bKc * LDB + (nBase + bRow));

  auto stage = [&](int buf) {
    // ---- A tile: 64 x 32, one 16B LDS store per thread ----
    {
      unsigned pk[4];
      if (aInB) {
        load8_bf(aPtr, pk);
      } else if (A_ONES && mg == M - 1) {
        pk[0] = pk[1] = pk[2] = pk[3] = 0x3F803F80u;   // bf16 1.0 pairs
      } else {
        pk[0] = pk[1] = pk[2] = pk[3] = 0u;
      }
      aPtr += 32;
      v4u w = {pk[0], pk[1], pk[2], pk[3]};
      *(v4u*)&As[buf][aRow][aKc] = w;
    }
    // ---- B tile: 32 x 64, stored transposed as Bs[n][k] ----
    {
      unsigned pk[4];
      if (B_TRANS) {
        // memory is Bsrc[n][k]: contiguous along k -> vector load
        load8_bf(bPtr, pk);
        bPtr += 32;
      } else {
        // memory is Bsrc[k][n]: 8 k's at constant offsets i*LDB (immediate offsets)
        unsigned short e[8];
        #pragma unroll
        for (int i = 0; i < 8; ++i)
          e[i] = elem_bf(bPtr + (long)i * LDB);
        bPtr += 32L * LDB;
        pk[0] = e[0] | ((unsigned)e[1] << 16);
        pk[1] = e[2] | ((unsigned)e[3] << 16);
        pk[2] = e[4] | ((unsigned)e[5] << 16);
        pk[3] = e[6] | ((unsigned)e[7] << 16);
      }
      if (B_RELU) {
        #pragma unroll
        for (int j = 0; j < 4; ++j) pk[j] = relu_pk(pk[j]);
      }
      v4u w = {pk[0], pk[1], pk[2], pk[3]};
      *(v4u*)&Bs[buf][bRow][bKc] = w;
    }
  };

  v8f acc0 = {};
  v8f acc1 = {};

  stage(0);
  int p = 0;
  for (int ks = 0; ks < K; ks += 32) {
    __syncthreads();
    if (ks + 32 < K) stage(p ^ 1);   // overlap next-tile staging with WMMA

    const v16bf af  = load_frag(&As[p][wm * 16 + lm][0], lh);
    const v16bf bf0 = load_frag(&Bs[p][wn * 32 + lm][0], lh);
    const v16bf bf1 = load_frag(&Bs[p][wn * 32 + 16 + lm][0], lh);
    acc0 = __builtin_amdgcn_wmma_f32_16x16x32_bf16(false, af, false, bf0,
                                                   (short)0, acc0, false, false);
    acc1 = __builtin_amdgcn_wmma_f32_16x16x32_bf16(false, af, false, bf1,
                                                   (short)0, acc1, false, false);
    p ^= 1;
  }

  // ---- epilogue (optionally fused BN); C lane mapping: n=lane&15, m=reg+8*(lane>>4)
  const int n0 = nBase + wn * 32 + lm;
  const int m0 = mBase + wm * 16 + lh * 8;
  #pragma unroll
  for (int r = 0; r < 8; ++r) {
    const int m = m0 + r;
    if (!M_GUARD || m < M) {
      float v0 = acc0[r], v1 = acc1[r];
      if (DO_BN) {
        const float sc = bn_g[m] * rsqrtf(bn_vr[m] + 1e-5f);
        const float sh = bn_b[m] - bn_mn[m] * sc;
        v0 = fmaf(v0, sc, sh);
        v1 = fmaf(v1, sc, sh);
      }
      store_c(Cp + (long)m * LDC + n0,      v0);
      store_c(Cp + (long)m * LDC + n0 + 16, v1);
    }
  }
  (void)N;
}

// Fused 5x5 depthwise (SAME) + grouped 8->8 pointwise conv, bf16 in/out, f32 math.
__global__ __launch_bounds__(256) void ms_fused_kernel(
    const unsigned short* __restrict__ qkv, const float* __restrict__ wdw,
    const float* __restrict__ wpw, unsigned short* __restrict__ ms2)
{
  const long idx = (long)blockIdx.x * blockDim.x + threadIdx.x; // 16*96*4096
  const int n = (int)(idx & 4095);
  const long tt = idx >> 12;
  const int g = (int)(tt % 96);
  const int b = (int)(tt / 96);
  const int h = n >> 6, w = n & 63;

  float dw[8];
  #pragma unroll
  for (int j = 0; j < 8; ++j) {
    const int c = g * 8 + j;
    const unsigned short* src = qkv + ((long)b * 768 + c) * 4096;
    const float* wt = wdw + c * 25;
    float s = 0.0f;
    #pragma unroll
    for (int dy = 0; dy < 5; ++dy) {
      const int hh = h + dy - 2;
      if (hh < 0 || hh > 63) continue;
      #pragma unroll
      for (int dx = 0; dx < 5; ++dx) {
        const int ww = w + dx - 2;
        if (ww < 0 || ww > 63) continue;
        s = fmaf(bf2f(src[hh * 64 + ww]), wt[dy * 5 + dx], s);
      }
    }
    dw[j] = s;
  }
  #pragma unroll
  for (int i = 0; i < 8; ++i) {
    const int o = g * 8 + i;
    const float* wp = wpw + o * 8;
    float s = 0.0f;
    #pragma unroll
    for (int j = 0; j < 8; ++j) s = fmaf(wp[j], dw[j], s);
    ms2[((long)b * 768 + o) * 4096 + n] = f2bf_bits(s);
  }
}

// attn[b, g*256+d, n] = raw[(b,g), d, n] / (raw[(b,g), 256, n] + 1e-15), fp32 -> bf16
__global__ __launch_bounds__(256) void attn_div_kernel(
    const float* __restrict__ raw, unsigned short* __restrict__ attn)
{
  const long idx = (long)blockIdx.x * blockDim.x + threadIdx.x; // 32*256*4096
  const int n = (int)(idx & 4095);
  const long tt = idx >> 12;
  const int d = (int)(tt & 255);
  const long bg = tt >> 8;                 // b*2+g
  const float* r = raw + bg * (long)257 * 4096;
  const float denom = r[(long)256 * 4096 + n] + 1e-15f;
  const float v = r[(long)d * 4096 + n];
  const int g = (int)(bg & 1), b = (int)(bg >> 1);
  attn[((long)b * 512 + g * 256 + d) * 4096 + n] = f2bf_bits(v / denom);
}

extern "C" void kernel_launch(void* const* d_in, const int* in_sizes, int n_in,
                              void* d_out, int out_size, void* d_ws, size_t ws_size,
                              hipStream_t stream) {
  (void)in_sizes; (void)n_in; (void)out_size; (void)ws_size;
  typedef unsigned short u16;

  const float* x     = (const float*)d_in[0];
  const float* w_qkv = (const float*)d_in[1];
  const float* w_dw  = (const float*)d_in[2];
  const float* w_pw  = (const float*)d_in[3];
  const float* w_o   = (const float*)d_in[4];
  const float* bn_g  = (const float*)d_in[5];
  const float* bn_b  = (const float*)d_in[6];
  const float* bn_m  = (const float*)d_in[7];
  const float* bn_v  = (const float*)d_in[8];
  float* out = (float*)d_out;

  char* ws = (char*)d_ws;
  u16*   qkv  = (u16*)(ws);                    // (16,768,4096) bf16
  u16*   ms2  = (u16*)(ws + 100663296L);       // (16,768,4096) bf16
  u16*   vk   = (u16*)(ws + 201326592L);       // (32,257,256)  bf16
  float* raw  = (float*)(ws + 205537280L);     // (32,257,4096) f32
  u16*   attn = (u16*)(ws + 340279296L);       // (16,512,4096) bf16

  // 1) qkv[b] = w_qkv (768x256) @ x[b] (256x4096), f32 in -> bf16 out
  wmma_gemm<float,float,u16, 768,4096,256, 256,4096,4096,
            false,false,false, false,false,false,false>
      <<<dim3(64,12,16),256,0,stream>>>(
      w_qkv, nullptr, 0L, 0L,
      x,     nullptr, (long)256*4096, 0L,
      qkv,   (long)768*4096,
      nullptr, nullptr, nullptr, nullptr);

  // 2) multiscale branch: fused depthwise 5x5 + grouped pointwise, bf16 -> bf16
  ms_fused_kernel<<<24576, 256, 0, stream>>>(qkv, w_dw, w_pw, ms2);

  // 3) vk[(b,g)] = v_aug (257x4096) @ relu(k)^T (4096x256)  [NT GEMM, bf16 -> bf16]
  wmma_gemm<u16,u16,u16, 257,256,4096, 4096,4096,256,
            true,true,true, true,true,true,false>
      <<<dim3(4,5,32),256,0,stream>>>(
      qkv, ms2, (long)768*4096, (long)512*4096,   // A: v channels 512..767 (+ones row)
      qkv, ms2, (long)768*4096, (long)256*4096,   // B: k channels 256..511, relu, B[k][n]=src[n*ldb+k]
      vk,  (long)257*256,
      nullptr, nullptr, nullptr, nullptr);

  // 4) raw[(b,g)] = vk (257x256) @ relu(q) (256x4096), bf16 -> f32
  wmma_gemm<u16,u16,float, 257,4096,256, 256,4096,4096,
            false,false,true, true,true,false,false>
      <<<dim3(64,5,32),256,0,stream>>>(
      vk,  nullptr, (long)257*256, 0L,
      qkv, ms2, (long)768*4096, 0L,               // B: q channels 0..255, relu
      raw, (long)257*4096,
      nullptr, nullptr, nullptr, nullptr);

  // 5) normalize by denominator row, f32 -> bf16
  attn_div_kernel<<<131072, 256, 0, stream>>>(raw, attn);

  // 6) out[b] = w_o (256x512) @ attn[b] (512x4096), fused BN, -> f32 d_out
  wmma_gemm<float,u16,float, 256,4096,512, 512,4096,4096,
            false,false,false, false,false,false,true>
      <<<dim3(64,4,16),256,0,stream>>>(
      w_o,  nullptr, 0L, 0L,
      attn, nullptr, (long)512*4096, 0L,
      out,  (long)256*4096,
      bn_g, bn_b, bn_m, bn_v);
}
